// SMM_UNet_14310831031045
// MI455X (gfx1250) — compile-verified, matrix operand order checked by reading, not run on hardware
//
#include <hip/hip_runtime.h>
#include <hip/hip_bf16.h>

// ---------------- problem constants ----------------
#define BSZ     16
#define C_IN_   64
#define HH      192
#define WW      192
#define L_SEQ   36864          // H*W
#define SEQS    64             // 4 * BSZ
#define DIN     32             // D_INNER
#define DST     16             // D_STATE
#define C_OUT_  48

#define NP      96             // partitions per sequence (parallel scan)
#define LP      384            // L_SEQ / NP
#define CT      64             // chunk length inside a partition
#define NCHUNK  6              // LP / CT

typedef float     v2f  __attribute__((ext_vector_type(2)));
typedef float     v8f  __attribute__((ext_vector_type(8)));
typedef _Float16  v16h __attribute__((ext_vector_type(16)));

// async-to-LDS builtin takes pointers to 4xint vectors in AS(1)/AS(3)
typedef int v4i_vs __attribute__((vector_size(4 * sizeof(int))));
typedef __attribute__((address_space(1))) v4i_vs* gptr_v4i;
typedef __attribute__((address_space(3))) v4i_vs* lptr_v4i;

// gfx1250 async global->LDS path (ASYNCcnt-tracked), with safe fallback
#if defined(__has_builtin)
#  if __has_builtin(__builtin_amdgcn_global_load_async_to_lds_b128) && \
      __has_builtin(__builtin_amdgcn_s_wait_asynccnt)
#    define USE_ASYNC_LDS 1
#  endif
#endif
#ifndef USE_ASYNC_LDS
#  define USE_ASYNC_LDS 0
#endif

__device__ __forceinline__ float siluf(float v) { return v / (1.f + __expf(-v)); }

// =====================================================================
// Kernel 1: layernorm over C=64, write xs (S,L,16) f32,
// input projection xz = u @ in_w.T via V_WMMA_F32_16X16X4_F32,
// store xz as bf16 (halves the dominant HBM term: written 1x, read 2x),
// accumulate sum_xs for the SE gate.
// Block = 256 threads = 8 waves, each wave owns a 16-position tile.
// =====================================================================
__global__ void __launch_bounds__(256)
k_ln_inproj(const float* __restrict__ x, const float* __restrict__ ng,
            const float* __restrict__ nb, const float* __restrict__ in_w,
            float* __restrict__ xs, __hip_bfloat16* __restrict__ xz,
            float* __restrict__ sum_xs)
{
    __shared__ float tile[8][16][65];
    __shared__ float mu_s[8][16], rs_s[8][16];
    __shared__ float inw_s[64 * 16];
    __shared__ float ngs[64], nbs[64];

    const int tid = threadIdx.x;
    for (int e = tid; e < 64 * 16; e += 256) inw_s[e] = in_w[e];
    if (tid < 64) { ngs[tid] = ng[tid]; nbs[tid] = nb[tid]; }

    const int wave = tid >> 5, lane = tid & 31;
    const int pblk = blockIdx.x * 128;              // 128 consecutive (b,l) positions
    const int b    = pblk / L_SEQ;                  // L_SEQ % 128 == 0 -> single b per block
    const int l0b  = pblk - b * L_SEQ;
    const int l0   = l0b + wave * 16;

    const float* xb = x + (size_t)b * C_IN_ * L_SEQ;
    #pragma unroll 4
    for (int r = 0; r < 32; ++r) {                  // 16 pos x 64 ch per wave
        int j = r * 32 + lane;
        int i = j & 15, c = j >> 4;
        tile[wave][i][c] = xb[(size_t)c * L_SEQ + l0 + i];
    }
    __syncthreads();

    if (tid < 128) {                                // per-position mean / rstd
        int w = tid >> 4, i = tid & 15;
        float s1 = 0.f, s2 = 0.f;
        for (int c = 0; c < 64; ++c) { float v = tile[w][i][c]; s1 += v; s2 += v * v; }
        float mu  = s1 * 0.015625f;
        float var = s2 * 0.015625f - mu * mu;
        mu_s[w][i] = mu;
        rs_s[w][i] = rsqrtf(var + 1e-5f);
    }
    __syncthreads();

    auto xn = [&](int w, int i, int c) -> float {
        return (tile[w][i][c] - mu_s[w][i]) * rs_s[w][i] * ngs[c] + nbs[c];
    };

    // store xs (S,L,16) with s = g*16 + b
    for (int e = tid; e < 8192; e += 256) {
        int w = e >> 10, rem = e & 1023, i = rem >> 6, c = rem & 63;
        int g = c >> 4, k = c & 15;
        size_t l = (size_t)(l0b + w * 16 + i);
        xs[(((size_t)(g * 16 + b)) * L_SEQ + l) * 16 + k] = xn(w, i, c);
    }
    // SE accumulator: sum over positions of xs, one atomic per (s,k) per block
    if (tid < 64) {
        int c = tid;
        float a = 0.f;
        for (int w = 0; w < 8; ++w)
            for (int i = 0; i < 16; ++i) a += xn(w, i, c);
        int g = c >> 4, k = c & 15;
        atomicAdd(&sum_xs[(g * 16 + b) * 16 + k], a);
    }

    // ---- WMMA input projection: per group g, xz_tile(16x64) = A(16x16) x in_w.T ----
    const int  Mrow = lane & 15;
    const bool hi   = lane >= 16;
    #pragma unroll
    for (int g = 0; g < 4; ++g) {
        v2f af[4];
        #pragma unroll
        for (int kc = 0; kc < 4; ++kc) {            // A 16x4 fragment (ISA layout)
            int k0 = kc * 4 + (hi ? 2 : 0);
            af[kc][0] = xn(wave, Mrow, g * 16 + k0);
            af[kc][1] = xn(wave, Mrow, g * 16 + k0 + 1);
        }
        #pragma unroll
        for (int nt = 0; nt < 4; ++nt) {
            v8f cf = {};
            #pragma unroll
            for (int kc = 0; kc < 4; ++kc) {        // B 4x16 fragment: B[k][n] = in_w[n][k]
                int k0 = kc * 4 + (hi ? 2 : 0);
                v2f bf;
                bf[0] = inw_s[(nt * 16 + Mrow) * 16 + k0];
                bf[1] = inw_s[(nt * 16 + Mrow) * 16 + k0 + 1];
                cf = __builtin_amdgcn_wmma_f32_16x16x4_f32(
                         false, af[kc], false, bf, (short)0, cf, false, false);
            }
            __hip_bfloat16* dst =
                xz + (((size_t)(g * 16 + b)) * L_SEQ + l0) * 64 + nt * 16;
            #pragma unroll
            for (int r = 0; r < 8; ++r) {           // C/D layout: lane<16 -> M=r, else M=8+r
                int M = hi ? (8 + r) : r;
                dst[(size_t)M * 64 + Mrow] = __float2bfloat16(cf[r]);
            }
        }
    }
}

// =====================================================================
// Scan kernels. 512 threads: tid -> (d = tid>>4, n = tid&15) state owner.
// PHASE 1: per-partition (prod a, h from h0=0).  PHASE 3: replay with
// corrected h0, emit m = ((y + D*x)*silu(z)) @ out_w.T, accumulate sum_m.
// Chunk staging uses GLOBAL_LOAD_ASYNC_TO_LDS_B128 when available.
// =====================================================================
template <int PHASE>
__global__ void __launch_bounds__(512)
k_scan(const __hip_bfloat16* __restrict__ xz,
       const float* __restrict__ conv_w, const float* __restrict__ conv_b,
       const float* __restrict__ xproj_w, const float* __restrict__ dtproj_w,
       const float* __restrict__ dtproj_b, const float* __restrict__ A_log,
       const float* __restrict__ Dvec, const float* __restrict__ out_w,
       float* __restrict__ prodA, float* __restrict__ h0buf,
       float* __restrict__ m_out, float* __restrict__ sum_m)
{
    __shared__ __align__(16) __hip_bfloat16 xzb[CT + 3][64];
    __shared__ float xcb[CT][32];
    __shared__ float xdblb[CT][33];
    __shared__ float dtl[CT][32];
    __shared__ float yb[CT][32];
    __shared__ float xprojw_s[33 * 32];
    __shared__ float convw_s[32 * 4], convb_s[32], dtw_s[32], dtb_s[32];
    __shared__ float An_s[32 * 16], Dv_s[32], outw_s[16 * 32];
    __shared__ float msum[16];

    const int tid = threadIdx.x;
    for (int e = tid; e < 33 * 32; e += 512) xprojw_s[e] = xproj_w[e];
    for (int e = tid; e < 32 * 16; e += 512) An_s[e] = -__expf(A_log[e]);
    for (int e = tid; e < 16 * 32; e += 512) outw_s[e] = out_w[e];
    if (tid < 128) convw_s[tid] = conv_w[tid];
    if (tid < 32) { convb_s[tid] = conv_b[tid]; dtw_s[tid] = dtproj_w[tid];
                    dtb_s[tid] = dtproj_b[tid]; Dv_s[tid] = Dvec[tid]; }
    if (tid < 16) msum[tid] = 0.f;
    __syncthreads();

    const int bid  = blockIdx.x;
    const int s    = bid / NP;
    const int pidx = bid - s * NP;
    const int tpart = pidx * LP;
    const int d = tid >> 4, n = tid & 15;
    const float An = An_s[d * 16 + n];

    float h, p = 1.f;
    if (PHASE == 1) h = 0.f;
    else            h = h0buf[(size_t)bid * 512 + tid];

    float sm0 = 0.f, sm1 = 0.f;
    const int tw = tid >> 3, j = tid & 7;

    for (int ch = 0; ch < NCHUNK; ++ch) {
        const int t0 = tpart + ch * CT;
        __syncthreads();
        // ---- stage A: stage xz chunk (with 3-step conv history) into LDS ----
        const size_t gelem = ((size_t)s * L_SEQ + t0 - 3) * 64;   // bf16 elements
#if USE_ASYNC_LDS
        {
            const unsigned ldsbase = (unsigned)(uintptr_t)&xzb[0][0];
            const char* gbytes = (const char*)xz + gelem * 2;
            // (CT+3)*64 bf16 = 536 x 16B transfers
            for (int e = tid; e < (CT + 3) * 8; e += 512) {
                if (t0 == 0 && e < 24) {              // rows t<0 at sequence start -> 0
                    unsigned* zz = (unsigned*)&xzb[0][0];
                    zz[e * 4 + 0] = 0u; zz[e * 4 + 1] = 0u;
                    zz[e * 4 + 2] = 0u; zz[e * 4 + 3] = 0u;
                } else {
                    __builtin_amdgcn_global_load_async_to_lds_b128(
                        (gptr_v4i)(uintptr_t)(gbytes + (size_t)e * 16),
                        (lptr_v4i)(uintptr_t)(ldsbase + (unsigned)e * 16u),
                        0, 0);
                }
            }
            __builtin_amdgcn_s_wait_asynccnt(0);
        }
#else
        {
            const unsigned* src32 = (const unsigned*)(xz + gelem);
            unsigned* dst32 = (unsigned*)&xzb[0][0];
            for (int e = tid; e < (CT + 3) * 32; e += 512) {
                int gt = t0 - 3 + (e >> 5);
                dst32[e] = (gt >= 0) ? src32[e] : 0u;
            }
        }
#endif
        if (ch + 1 < NCHUNK) {      // prefetch next chunk -> global_prefetch_b8
            const __hip_bfloat16* nxt = xz + ((size_t)s * L_SEQ + t0 + CT - 3) * 64;
            if (tid < CT + 3) __builtin_prefetch(nxt + tid * 64, 0, 1);
        }
        __syncthreads();
        // ---- stage B: depthwise causal conv + silu ----
        {
            int t = tw;
            #pragma unroll
            for (int q = 0; q < 4; ++q) {
                int dd = j + q * 8;
                float acc = convb_s[dd];
                #pragma unroll
                for (int kk = 0; kk < 4; ++kk)
                    acc += convw_s[dd * 4 + kk] * __bfloat162float(xzb[t + kk][dd]);
                xcb[t][dd] = siluf(acc);
            }
        }
        __syncthreads();
        // ---- stage C: xdbl = xc @ xproj_w.T (33 outputs) ----
        {
            int t = tw;
            for (int r = j; r < 33; r += 8) {
                float acc = 0.f;
                #pragma unroll
                for (int dd = 0; dd < 32; ++dd) acc += xcb[t][dd] * xprojw_s[r * 32 + dd];
                xdblb[t][r] = acc;
            }
        }
        __syncthreads();
        // ---- stage D: dt = softplus(xdbl0 * dtw + dtb) ----
        {
            int t = tw;
            float x0 = xdblb[t][0];
            #pragma unroll
            for (int q = 0; q < 4; ++q) {
                int dd = j + q * 8;
                float v = x0 * dtw_s[dd] + dtb_s[dd];
                dtl[t][dd] = (v > 20.f) ? v : log1pf(__expf(v));
            }
        }
        __syncthreads();
        // ---- sequential recurrence over the chunk ----
        for (int t = 0; t < CT; ++t) {
            float dtv = dtl[t][d];
            float a   = __expf(dtv * An);
            h = a * h + dtv * xdblb[t][1 + n] * xcb[t][d];
            if (PHASE == 1) {
                p *= a;
            } else {
                float y = h * xdblb[t][17 + n];       // C_t[n]
                y += __shfl_xor(y, 1);
                y += __shfl_xor(y, 2);
                y += __shfl_xor(y, 4);
                y += __shfl_xor(y, 8);                // reduce over n (16-lane group)
                if (n == 0) yb[t][d] = y;
            }
        }
        if (PHASE == 3) {
            __syncthreads();
            { // yfin = (y + D*x) * silu(z)
                int t = tw;
                #pragma unroll
                for (int q = 0; q < 4; ++q) {
                    int dd = j + q * 8;
                    float z = __bfloat162float(xzb[t + 3][32 + dd]);
                    yb[t][dd] = (yb[t][dd] + Dv_s[dd] * xcb[t][dd]) * siluf(z);
                }
            }
            __syncthreads();
            { // m = yfin @ out_w.T (16 outputs)
                int t = tw;
                float* mrow = m_out + ((size_t)s * L_SEQ + t0 + t) * 16;
                #pragma unroll
                for (int q = 0; q < 2; ++q) {
                    int o = j + q * 8;
                    float acc = 0.f;
                    #pragma unroll
                    for (int dd = 0; dd < 32; ++dd) acc += yb[t][dd] * outw_s[o * 32 + dd];
                    mrow[o] = acc;
                    if (q == 0) sm0 += acc; else sm1 += acc;
                }
            }
        }
    }

    if (PHASE == 1) {
        prodA[(size_t)bid * 512 + tid] = p;
        h0buf[(size_t)bid * 512 + tid] = h;
    } else {
        __syncthreads();
        atomicAdd(&msum[j],     sm0);
        atomicAdd(&msum[j + 8], sm1);
        __syncthreads();
        if (tid < 16) atomicAdd(&sum_m[s * 16 + tid], msum[tid]);
    }
}

// Phase 2: sequential prefix over partitions; rewrite h0buf with the
// corrected initial state of each partition.  64 blocks x 512 threads.
__global__ void __launch_bounds__(512)
k_scan_fix(float* __restrict__ prodA, float* __restrict__ h0buf)
{
    const int s = blockIdx.x, tid = threadIdx.x;
    float carry = 0.f;
    for (int pu = 0; pu < NP; ++pu) {
        size_t idx = ((size_t)(s * NP + pu)) * 512 + tid;
        float a  = prodA[idx];
        float he = h0buf[idx];
        h0buf[idx] = carry;
        carry = a * carry + he;
    }
}

// SE gate: avg -> relu(avg @ se_w1.T) @ se_w2.T -> sigmoid. 64 threads.
__global__ void k_gate(const float* __restrict__ sum_m, const float* __restrict__ sum_xs,
                       const float* __restrict__ skip, const float* __restrict__ se_w1,
                       const float* __restrict__ se_w2, float* __restrict__ gate)
{
    int s = threadIdx.x;
    if (s >= SEQS) return;
    const float inv = 1.f / (float)L_SEQ;
    const float sk  = skip[0];
    float avg[32];
    for (int k = 0; k < 16; ++k) {
        avg[k]      = sum_m[s * 16 + k] * inv;
        avg[16 + k] = sk * sum_xs[s * 16 + k] * inv;
    }
    float h0 = 0.f, h1 = 0.f;
    for (int k = 0; k < 32; ++k) { h0 += avg[k] * se_w1[k]; h1 += avg[k] * se_w1[32 + k]; }
    h0 = fmaxf(h0, 0.f); h1 = fmaxf(h1, 0.f);
    for (int jj = 0; jj < 32; ++jj) {
        float v = h0 * se_w2[jj * 2 + 0] + h1 * se_w2[jj * 2 + 1];
        gate[s * 32 + jj] = 1.f / (1.f + __expf(-v));
    }
}

// =====================================================================
// Kernel 5: out_c = g1*m + g2*skip*xs, layernorm over C=64, then
// 64->48 projection via V_WMMA_F32_16X16X32_F16 (+bias), write NCHW.
// =====================================================================
__global__ void __launch_bounds__(256)
k_out(const float* __restrict__ m_in, const float* __restrict__ xs,
      const float* __restrict__ gate, const float* __restrict__ skip,
      const float* __restrict__ ng, const float* __restrict__ nb,
      const float* __restrict__ proj_w, const float* __restrict__ proj_b,
      float* __restrict__ out)
{
    __shared__ float tile[8][16][65];
    __shared__ float mu_s[8][16], rs_s[8][16];
    __shared__ float pw_s[48 * 64];
    __shared__ float pb_s[48], ngs[64], nbs[64], gt_s[128];

    const int tid  = threadIdx.x;
    const int pblk = blockIdx.x * 128;
    const int b    = pblk / L_SEQ;
    const int l0b  = pblk - b * L_SEQ;

    for (int e = tid; e < 48 * 64; e += 256) pw_s[e] = proj_w[e];
    if (tid < 48) pb_s[tid] = proj_b[tid];
    if (tid < 64) { ngs[tid] = ng[tid]; nbs[tid] = nb[tid]; }
    if (tid < 128) { int g = tid >> 5, k = tid & 31; gt_s[tid] = gate[(g * 16 + b) * 32 + k]; }
    const float sk = skip[0];

    for (int e = tid; e < 8192; e += 256) {     // gated combine into LDS tile
        int w = e >> 10, rem = e & 1023, i = rem >> 6, c = rem & 63;
        int g = c >> 4, k = c & 15;
        size_t base = (((size_t)(g * 16 + b)) * L_SEQ + (l0b + w * 16 + i)) * 16 + k;
        tile[w][i][c] = gt_s[g * 32 + k] * m_in[base] + gt_s[g * 32 + 16 + k] * sk * xs[base];
    }
    __syncthreads();
    if (tid < 128) {
        int w = tid >> 4, i = tid & 15;
        float s1 = 0.f, s2 = 0.f;
        for (int c = 0; c < 64; ++c) { float v = tile[w][i][c]; s1 += v; s2 += v * v; }
        float mu  = s1 * 0.015625f;
        float var = s2 * 0.015625f - mu * mu;
        mu_s[w][i] = mu;
        rs_s[w][i] = rsqrtf(var + 1e-5f);
    }
    __syncthreads();

    const int  wave = tid >> 5, lane = tid & 31;
    const int  Mrow = lane & 15;
    const bool hi   = lane >= 16;
    const int  l0   = l0b + wave * 16;
    auto xn = [&](int i, int c) -> float {
        return (tile[wave][i][c] - mu_s[wave][i]) * rs_s[wave][i] * ngs[c] + nbs[c];
    };

    // A fragments: 16x64 split into two 16x32 f16 fragments (ISA 16-bit A layout)
    v16h afr[2];
    #pragma unroll
    for (int kc = 0; kc < 2; ++kc) {
        int koff = kc * 32, kbase = hi ? 8 : 0;
        #pragma unroll
        for (int v = 0; v < 8; ++v) {
            int k0 = (v < 4) ? (kbase + 2 * v) : (16 + kbase + 2 * (v - 4));
            afr[kc][2 * v]     = (_Float16)xn(Mrow, koff + k0);
            afr[kc][2 * v + 1] = (_Float16)xn(Mrow, koff + k0 + 1);
        }
    }
    #pragma unroll
    for (int nt = 0; nt < 3; ++nt) {            // 48 outputs = 3 N-tiles
        v8f cf = {};
        #pragma unroll
        for (int kc = 0; kc < 2; ++kc) {
            int koff = kc * 32, kbase = hi ? 8 : 0;
            v16h bf;                             // B[k][n] = proj_w[n][k]
            #pragma unroll
            for (int v = 0; v < 8; ++v) {
                int k0 = (v < 4) ? (kbase + 2 * v) : (16 + kbase + 2 * (v - 4));
                bf[2 * v]     = (_Float16)pw_s[(nt * 16 + Mrow) * 64 + koff + k0];
                bf[2 * v + 1] = (_Float16)pw_s[(nt * 16 + Mrow) * 64 + koff + k0 + 1];
            }
            cf = __builtin_amdgcn_wmma_f32_16x16x32_f16(
                     false, afr[kc], false, bf, (short)0, cf, false, false);
        }
        #pragma unroll
        for (int r = 0; r < 8; ++r) {
            int M = hi ? (8 + r) : r;
            int N = nt * 16 + Mrow;
            out[((size_t)b * C_OUT_ + N) * L_SEQ + l0 + M] = cf[r] + pb_s[N];
        }
    }
}

// =====================================================================
extern "C" void kernel_launch(void* const* d_in, const int* in_sizes, int n_in,
                              void* d_out, int out_size, void* d_ws, size_t ws_size,
                              hipStream_t stream)
{
    const float* x        = (const float*)d_in[0];
    const float* norm_g   = (const float*)d_in[1];
    const float* norm_b   = (const float*)d_in[2];
    const float* skip_sc  = (const float*)d_in[3];
    const float* se_w1    = (const float*)d_in[4];
    const float* se_w2    = (const float*)d_in[5];
    const float* proj_w   = (const float*)d_in[6];
    const float* proj_b   = (const float*)d_in[7];
    const float* in_w     = (const float*)d_in[8];
    const float* conv_w   = (const float*)d_in[9];
    const float* conv_b   = (const float*)d_in[10];
    const float* xproj_w  = (const float*)d_in[11];
    const float* dtproj_w = (const float*)d_in[12];
    const float* dtproj_b = (const float*)d_in[13];
    const float* A_log    = (const float*)d_in[14];
    const float* Dvec     = (const float*)d_in[15];
    const float* out_w    = (const float*)d_in[16];

    char* base = (char*)d_ws;
    float* xs = (float*)base;               base += (size_t)SEQS * L_SEQ * 16 * 4; // 151 MB
    __hip_bfloat16* xz = (__hip_bfloat16*)base;
                                            base += (size_t)SEQS * L_SEQ * 64 * 2; // 302 MB
    float* m_buf  = (float*)base;           base += (size_t)SEQS * L_SEQ * 16 * 4; // 151 MB
    float* prodA  = (float*)base;           base += (size_t)SEQS * NP * 512 * 4;   // 12.6 MB
    float* h0buf  = (float*)base;           base += (size_t)SEQS * NP * 512 * 4;   // 12.6 MB
    float* sum_m  = (float*)base;           base += SEQS * 16 * 4;
    float* sum_xs = (float*)base;           base += SEQS * 16 * 4;
    float* gate   = (float*)base;           base += SEQS * 32 * 4;

    // zero the SE accumulators (sum_m, sum_xs contiguous) — capturable async memset
    (void)hipMemsetAsync(sum_m, 0, 2 * SEQS * 16 * sizeof(float), stream);

    const int pos_blocks = (BSZ * L_SEQ) / 128;               // 4608

    k_ln_inproj<<<pos_blocks, 256, 0, stream>>>(x, norm_g, norm_b, in_w, xs, xz, sum_xs);

    k_scan<1><<<SEQS * NP, 512, 0, stream>>>(xz, conv_w, conv_b, xproj_w, dtproj_w,
                                             dtproj_b, A_log, Dvec, out_w,
                                             prodA, h0buf, m_buf, sum_m);
    k_scan_fix<<<SEQS, 512, 0, stream>>>(prodA, h0buf);
    k_scan<3><<<SEQS * NP, 512, 0, stream>>>(xz, conv_w, conv_b, xproj_w, dtproj_w,
                                             dtproj_b, A_log, Dvec, out_w,
                                             prodA, h0buf, m_buf, sum_m);

    k_gate<<<1, 64, 0, stream>>>(sum_m, sum_xs, skip_sc, se_w1, se_w2, gate);

    k_out<<<pos_blocks, 256, 0, stream>>>(m_buf, xs, gate, skip_sc, norm_g, norm_b,
                                          proj_w, proj_b, (float*)d_out);
}
